// MwAN_snli_64742337020402
// MI455X (gfx1250) — compile-verified, matrix operand order checked by reading, not run on hardware
//
#include <hip/hip_runtime.h>
#include <hip/hip_bf16.h>
#include <math.h>

// ---------------------------------------------------------------------------
// Model dims (fixed by the reference)
// ---------------------------------------------------------------------------
#define B_ 8
#define N_ 128
#define M_ 128
#define E_ 300
#define H_ 150
#define G_ 450        // 3*H
#define D2_ 300       // 2*H
#define AGG_ 1800     // 12*H
#define T_ 128

// ---------------------------------------------------------------------------
// CDNA5 WMMA types
// ---------------------------------------------------------------------------
typedef __attribute__((ext_vector_type(16))) __bf16 v16bf;
typedef __attribute__((ext_vector_type(8)))  float  v8f;

__device__ __forceinline__ unsigned short f32_to_bf16(float f) {
    unsigned u = __float_as_uint(f);
    unsigned r = u + 0x7FFFu + ((u >> 16) & 1u);   // round-to-nearest-even
    return (unsigned short)(r >> 16);
}
__device__ __forceinline__ float bf16_to_f32(unsigned short s) {
    return __uint_as_float(((unsigned)s) << 16);
}
__device__ __forceinline__ __bf16 bfbits(unsigned short s) {
    __bf16 b; __builtin_memcpy(&b, &s, 2); return b;
}
__device__ __forceinline__ __bf16 f2bf(float f) {
    return bfbits(f32_to_bf16(f));
}
__device__ __forceinline__ float sigmoidf_(float x) {
    return 1.0f / (1.0f + __expf(-x));
}

// Async DMA: global -> LDS, 16B granule, tracked by ASYNCcnt (CDNA5).
__device__ __forceinline__ void async_copy_b128(unsigned ldsByteAddr, const float* gptr) {
    unsigned long long ga = (unsigned long long)(size_t)gptr;
    asm volatile("global_load_async_to_lds_b128 %0, %1, off"
                 :: "v"(ldsByteAddr), "v"(ga) : "memory");
}
__device__ __forceinline__ void wait_async0() {
    asm volatile("s_wait_asynccnt 0x0" ::: "memory");
}

// 16-bit A/B fragment K-index map for V_WMMA 16x16x32 (wave32):
//   lane = half*16 + (m|n); element i of v16bf -> k:
//   i<8 : k = half*8 + i      (K 0..15 region)
//   i>=8: k = 8 + half*8 + i  (K 16..31 region)
__device__ __forceinline__ int kmap(int half, int i) {
    return (i < 8) ? (half * 8 + i) : (8 + half * 8 + i);
}

// ---------------------------------------------------------------------------
// K1: embedding gather  out[r,:] = emb[idx[r],:]
// ---------------------------------------------------------------------------
__global__ void gather_rows(const int* __restrict__ idx, const float* __restrict__ emb,
                            float* __restrict__ out, int rows, int width) {
    int r = blockIdx.x;
    if (r >= rows) return;
    const float* src = emb + (long long)idx[r] * width;
    float* dst = out + (long long)r * width;
    for (int i = threadIdx.x; i < width; i += blockDim.x) dst[i] = src[i];
}

// ---------------------------------------------------------------------------
// K2: generic tiled bf16 WMMA GEMM:  C[b] = A[b] (MxK) * Bw[b] (NxK)^T + bias
// Block tile 128x64; 8 waves in a 4x2 grid, each wave owns a 2x2 register
// tile of 16x16 WMMA accumulators (4 wmma per k-step, A/B fragments reused 2x).
// Interior tiles are staged fp32 into LDS with GLOBAL_LOAD_ASYNC_TO_LDS_B128
// (ASYNCcnt); edge tiles use branchless clamped scalar loads. f32->bf16
// conversion happens at fragment build (co-issues with the XDL WMMA pipe).
// LDS row stride 36 floats: 16B-aligned rows for B128 async + 4-bank stagger.
// ---------------------------------------------------------------------------
__global__ void wmma_gemm_bias(const float* __restrict__ A, long long asb, int lda,
                               const float* __restrict__ Bw, long long bsb, int ldb,
                               const float* __restrict__ bias,
                               float* __restrict__ C, long long csb, int ldc,
                               int M, int N, int K) {
    __shared__ alignas(16) float Atf[128][36];
    __shared__ alignas(16) float Btf[64][36];

    int bz = blockIdx.z;
    const float* Ab = A + (long long)bz * asb;
    const float* Bb = Bw + (long long)bz * bsb;
    float* Cb = C + (long long)bz * csb;

    int m0 = blockIdx.x * 128, n0 = blockIdx.y * 64;
    int tid = threadIdx.x;
    int lane = tid & 31, wv = tid >> 5;
    int rw = wv & 3, cw = wv >> 2;      // 4x2 wave grid; wave tile 32x32
    int half = lane >> 4, ln16 = lane & 15;

    v8f zero = {0.f, 0.f, 0.f, 0.f, 0.f, 0.f, 0.f, 0.f};
    v8f acc00 = zero, acc01 = zero, acc10 = zero, acc11 = zero;

    for (int k0 = 0; k0 < K; k0 += 32) {
        bool full = (m0 + 128 <= M) && (n0 + 64 <= N) && (k0 + 32 <= K);
        if (full) {
            // Async DMA path: A tile 128x32 = 1024 x 16B chunks, B tile 64x32 = 512.
            for (int i = tid; i < 1024; i += 256) {
                int r = i >> 3, c4 = (i & 7) * 4;
                async_copy_b128((unsigned)(size_t)&Atf[r][c4],
                                Ab + (long long)(m0 + r) * lda + (k0 + c4));
            }
            for (int i = tid; i < 512; i += 256) {
                int r = i >> 3, c4 = (i & 7) * 4;
                async_copy_b128((unsigned)(size_t)&Btf[r][c4],
                                Bb + (long long)(n0 + r) * ldb + (k0 + c4));
            }
            wait_async0();
        } else {
            // Edge path: branchless clamped loads + cndmask zero-fill.
            for (int i = tid; i < 128 * 32; i += 256) {
                int r = i >> 5, c = i & 31;
                int gm = m0 + r, gk = k0 + c;
                int cm = (gm < M) ? gm : (M - 1);
                int ck = (gk < K) ? gk : (K - 1);
                float v = Ab[(long long)cm * lda + ck];
                Atf[r][c] = (gm < M && gk < K) ? v : 0.f;
            }
            for (int i = tid; i < 64 * 32; i += 256) {
                int r = i >> 5, c = i & 31;
                int gn = n0 + r, gk = k0 + c;
                int cn = (gn < N) ? gn : (N - 1);
                int ck = (gk < K) ? gk : (K - 1);
                float v = Bb[(long long)cn * ldb + ck];
                Btf[r][c] = (gn < N && gk < K) ? v : 0.f;
            }
        }
        __syncthreads();

        v16bf a0, a1, b0, b1;
        int am0 = rw * 32 + ln16, am1 = am0 + 16;
        int bn0 = cw * 32 + ln16, bn1 = bn0 + 16;
#pragma unroll
        for (int i = 0; i < 16; ++i) {
            int kk = kmap(half, i);
            a0[i] = f2bf(Atf[am0][kk]);
            a1[i] = f2bf(Atf[am1][kk]);
            b0[i] = f2bf(Btf[bn0][kk]);
            b1[i] = f2bf(Btf[bn1][kk]);
        }
        acc00 = __builtin_amdgcn_wmma_f32_16x16x32_bf16(false, a0, false, b0, (short)0, acc00, false, false);
        acc01 = __builtin_amdgcn_wmma_f32_16x16x32_bf16(false, a0, false, b1, (short)0, acc01, false, false);
        acc10 = __builtin_amdgcn_wmma_f32_16x16x32_bf16(false, a1, false, b0, (short)0, acc10, false, false);
        acc11 = __builtin_amdgcn_wmma_f32_16x16x32_bf16(false, a1, false, b1, (short)0, acc11, false, false);
        __syncthreads();
    }

    // write 2x2 register tile; C layout: VGPR r -> row r + 8*half, col = lane&15
#pragma unroll
    for (int mt = 0; mt < 2; ++mt) {
#pragma unroll
        for (int nt = 0; nt < 2; ++nt) {
            v8f acc = (mt == 0) ? (nt == 0 ? acc00 : acc01) : (nt == 0 ? acc10 : acc11);
            int cn = n0 + cw * 32 + nt * 16 + ln16;
            if (cn < N) {
                float bv = bias ? bias[cn] : 0.f;
#pragma unroll
                for (int r = 0; r < 8; ++r) {
                    int cm = m0 + rw * 32 + mt * 16 + r + 8 * half;
                    if (cm < M) Cb[(long long)cm * ldc + cn] = acc[r] + bv;
                }
            }
        }
    }
}

// ---------------------------------------------------------------------------
// K3: persistent GRU scan, one block per (batch, direction).
// Whh (450x150) held in LDS as bf16 (135 KB of the 320 KB WGP LDS).
// gx already contains x@Wih^T + bih.  dir=1 runs time-reversed in place.
// Inner GEMV reads packed bf16 pairs (rows are 4B-aligned: 150*2B = 300B).
// ---------------------------------------------------------------------------
__global__ void gru_scan(const float* __restrict__ gx0, const float* __restrict__ gx1,
                         const float* __restrict__ Whh2, const float* __restrict__ bhh2,
                         float* __restrict__ Hs, int outStride) {
    extern __shared__ unsigned char smem[];
    unsigned short* wsh = (unsigned short*)smem;                 // 450*150 bf16
    float* fbase = (float*)(smem + (size_t)G_ * H_ * 2);
    float* hcur = fbase;            // 152
    float* hnew = fbase + 152;      // 152
    float* gh   = fbase + 304;      // 456

    int b = blockIdx.x, dir = blockIdx.y;
    const float* gx  = dir ? gx1 : gx0;
    const float* Whh = Whh2 + (long long)dir * G_ * H_;
    const float* bhh = bhh2 + dir * G_;

    for (int i = threadIdx.x; i < G_ * H_; i += blockDim.x) wsh[i] = f32_to_bf16(Whh[i]);
    for (int i = threadIdx.x; i < H_; i += blockDim.x) hcur[i] = 0.f;
    __syncthreads();

    for (int s = 0; s < T_; ++s) {
        int t = dir ? (T_ - 1 - s) : s;
        const float* gxt = gx + ((long long)b * T_ + t) * G_;

        for (int o = threadIdx.x; o < G_; o += blockDim.x) {
            float a = bhh[o];
            const unsigned int* wr = (const unsigned int*)(wsh + o * H_);
            for (int k2 = 0; k2 < H_ / 2; ++k2) {
                unsigned p = wr[k2];
                a += bf16_to_f32((unsigned short)(p & 0xFFFFu)) * hcur[2 * k2];
                a += bf16_to_f32((unsigned short)(p >> 16))     * hcur[2 * k2 + 1];
            }
            gh[o] = a;
        }
        __syncthreads();

        for (int j = threadIdx.x; j < H_; j += blockDim.x) {
            float r  = sigmoidf_(gxt[j]        + gh[j]);
            float z  = sigmoidf_(gxt[H_ + j]   + gh[H_ + j]);
            float nn = tanhf    (gxt[2*H_ + j] + r * gh[2*H_ + j]);
            float h  = (1.f - z) * nn + z * hcur[j];
            hnew[j] = h;
            Hs[((long long)b * T_ + t) * outStride + dir * H_ + j] = h;
        }
        __syncthreads();
        for (int j = threadIdx.x; j < H_; j += blockDim.x) hcur[j] = hnew[j];
        __syncthreads();
    }
}

// ---------------------------------------------------------------------------
// K4: additive attention scores  S[b,m,n] = sum_h v[h]*tanh(a[b,n,h] + sign*c[b,m,h])
// ---------------------------------------------------------------------------
__global__ void add_attn_scores(const float* __restrict__ Aa,   // (B,N,H)
                                const float* __restrict__ Cc,   // (B,M,H)
                                const float* __restrict__ v, float sign,
                                float* __restrict__ S) {
    __shared__ float ch[160], vh[160];
    int b = blockIdx.y, m = blockIdx.x;
    for (int i = threadIdx.x; i < H_; i += blockDim.x) {
        ch[i] = Cc[((long long)b * M_ + m) * H_ + i];
        vh[i] = v[i];
    }
    __syncthreads();
    for (int n = threadIdx.x; n < N_; n += blockDim.x) {
        const float* ar = Aa + ((long long)b * N_ + n) * H_;
        float s = 0.f;
        for (int h = 0; h < H_; ++h) s += vh[h] * tanhf(ar[h] + sign * ch[h]);
        S[((long long)b * M_ + m) * N_ + n] = s;
    }
}

// ---------------------------------------------------------------------------
// K7: trilinear attention scores via per-(b,m) WMMA GEMM:
//   S[b,m,n] = sum_h v[h] * tanh( sum_d Xn[b,n,d]*Ym[b,m,d]*W[h,d] )
// GEMM: (W ⊙ Ym_row) [160x300, bf16] x Xn[b]^T [300x128] with 16x16x32 WMMA,
// then tanh·v reduction across the accumulator rows (shfl_xor 16 folds halves).
// Branchless clamped loads in the tile loaders.
// ---------------------------------------------------------------------------
__global__ void trilinear_scores(const float* __restrict__ Xn,  // (B,N,300)
                                 const float* __restrict__ Ym,  // (B,M,300)
                                 const float* __restrict__ W,   // (150,300)
                                 const float* __restrict__ v,   // (150)
                                 float* __restrict__ S) {       // (B,M,N)
    __shared__ float u[304];
    __shared__ float vsh[160];
    __shared__ unsigned short At[160][33];
    __shared__ unsigned short Bt[32][129];

    int b = blockIdx.y, m = blockIdx.x;
    int tid = threadIdx.x, lane = tid & 31, wv = tid >> 5;
    int half = lane >> 4, ln16 = lane & 15;

    const float* ym = Ym + ((long long)b * M_ + m) * D2_;
    for (int i = tid; i < D2_; i += 256) u[i] = ym[i];
    for (int i = tid; i < 160; i += 256) vsh[i] = (i < H_) ? v[i] : 0.f;
    __syncthreads();

    v8f zero = {0.f, 0.f, 0.f, 0.f, 0.f, 0.f, 0.f, 0.f};
    v8f acc[10];
#pragma unroll
    for (int i = 0; i < 10; ++i) acc[i] = zero;

    for (int k0 = 0; k0 < D2_; k0 += 32) {
        for (int i = tid; i < 160 * 32; i += 256) {
            int r = i >> 5, c = i & 31;
            int gk = k0 + c;
            int cr = (r < H_) ? r : (H_ - 1);
            int ck = (gk < D2_) ? gk : (D2_ - 1);
            float val = W[(long long)cr * D2_ + ck] * u[ck];
            val = (r < H_ && gk < D2_) ? val : 0.f;
            At[r][c] = f32_to_bf16(val);
        }
        for (int i = tid; i < 32 * 128; i += 256) {
            int r = i >> 7, c = i & 127;       // r: k within tile, c: n
            int gk = k0 + r;
            int ck = (gk < D2_) ? gk : (D2_ - 1);
            float val = Xn[((long long)b * N_ + c) * D2_ + ck];
            val = (gk < D2_) ? val : 0.f;
            Bt[r][c] = f32_to_bf16(val);
        }
        __syncthreads();

        v16bf bfv;
        int n = wv * 16 + ln16;                // wave owns one 16-col tile of n
#pragma unroll
        for (int i = 0; i < 16; ++i) bfv[i] = bfbits(Bt[kmap(half, i)][n]);

#pragma unroll
        for (int rt = 0; rt < 10; ++rt) {
            v16bf af;
            int hrow = rt * 16 + ln16;
#pragma unroll
            for (int i = 0; i < 16; ++i) af[i] = bfbits(At[hrow][kmap(half, i)]);
            acc[rt] = __builtin_amdgcn_wmma_f32_16x16x32_bf16(false, af, false, bfv,
                                                              (short)0, acc[rt], false, false);
        }
        __syncthreads();
    }

    float s = 0.f;
#pragma unroll
    for (int rt = 0; rt < 10; ++rt) {
#pragma unroll
        for (int r = 0; r < 8; ++r) {
            int h = rt * 16 + r + 8 * half;
            s += tanhf(acc[rt][r]) * vsh[h];   // vsh==0 for padded h>=150
        }
    }
    s += __shfl_xor(s, 16, 32);                // fold the two row-halves
    if (lane < 16) {
        int n = wv * 16 + lane;
        S[((long long)b * M_ + m) * N_ + n] = s;
    }
}

// ---------------------------------------------------------------------------
// K5: softmax over last axis of S[b,m,:n] then attend over values (B,n,Dv).
// Out[b*obs + m*ors + off + d]
// ---------------------------------------------------------------------------
__global__ void softmax_attend(const float* __restrict__ S,
                               const float* __restrict__ Vv, int n, int Dv,
                               float* __restrict__ Out,
                               long long outRowStride, long long outBatchStride, int outOff) {
    __shared__ float w[N_];
    __shared__ float red[1];
    int b = blockIdx.y, m = blockIdx.x;
    const float* srow = S + ((long long)b * gridDim.x + m) * n;
    for (int i = threadIdx.x; i < n; i += blockDim.x) w[i] = srow[i];
    __syncthreads();
    if (threadIdx.x == 0) {
        float mx = -3.4e38f;
        for (int i = 0; i < n; ++i) mx = fmaxf(mx, w[i]);
        float sm = 0.f;
        for (int i = 0; i < n; ++i) { w[i] = __expf(w[i] - mx); sm += w[i]; }
        red[0] = sm;
    }
    __syncthreads();
    float inv = 1.f / red[0];
    float* orow = Out + (long long)b * outBatchStride + (long long)m * outRowStride + outOff;
    for (int d = threadIdx.x; d < Dv; d += blockDim.x) {
        float a = 0.f;
        for (int i = 0; i < n; ++i) a += w[i] * Vv[((long long)b * n + i) * Dv + d];
        orow[d] = a * inv;
    }
}

// ---------------------------------------------------------------------------
// Small glue kernels
// ---------------------------------------------------------------------------
__global__ void copy_to_agg(const float* __restrict__ src, float* __restrict__ agg,
                            long long total, int width, int aggStride, int off) {
    long long i = (long long)blockIdx.x * blockDim.x + threadIdx.x;
    if (i < total) {
        long long r = i / width; int d = (int)(i % width);
        agg[r * aggStride + off + d] = src[i];
    }
}

__global__ void tanh_dot_v(const float* __restrict__ A, const float* __restrict__ v,
                           float* __restrict__ out, int rows) {
    int r = blockIdx.x * blockDim.x + threadIdx.x;
    if (r < rows) {
        const float* a = A + (long long)r * H_;
        float s = 0.f;
        for (int h = 0; h < H_; ++h) s += tanhf(a[h]) * v[h];
        out[r] = s;
    }
}

__global__ void sp_scores(const float* __restrict__ t1,  // (B,M,H)
                          const float* __restrict__ rqp, // (B,H)
                          const float* __restrict__ vp,
                          float* __restrict__ sp) {      // (B,M)
    int i = blockIdx.x * blockDim.x + threadIdx.x;
    if (i < B_ * M_) {
        int b = i >> 7;
        float s = 0.f;
        for (int h = 0; h < H_; ++h) s += (t1[(long long)i * H_ + h] + rqp[b * H_ + h]) * vp[h];
        sp[i] = s;
    }
}

__global__ void pred_kernel(const float* __restrict__ rp, const float* __restrict__ Wpred,
                            float* __restrict__ out) {
    int b = blockIdx.x, c = threadIdx.x;
    if (c < 3) {
        float s = 0.f;
        for (int k = 0; k < D2_; ++k) s += rp[b * D2_ + k] * Wpred[c * D2_ + k];
        out[b * 3 + c] = (s > 0.f) ? s : 0.01f * s;   // leaky_relu, slope 0.01
    }
}

// ---------------------------------------------------------------------------
// Orchestration
// ---------------------------------------------------------------------------
extern "C" void kernel_launch(void* const* d_in, const int* in_sizes, int n_in,
                              void* d_out, int out_size, void* d_ws, size_t ws_size,
                              hipStream_t stream) {
    const int*   premise    = (const int*)d_in[0];
    const int*   hypothesis = (const int*)d_in[1];
    const float* emb        = (const float*)d_in[2];
    const float* prem_Wih = (const float*)d_in[3];
    const float* prem_Whh = (const float*)d_in[4];
    const float* prem_bih = (const float*)d_in[5];
    const float* prem_bhh = (const float*)d_in[6];
    const float* hypo_Wih = (const float*)d_in[7];
    const float* hypo_Whh = (const float*)d_in[8];
    const float* hypo_bih = (const float*)d_in[9];
    const float* hypo_bhh = (const float*)d_in[10];
    const float* agg_Wih  = (const float*)d_in[11];
    const float* agg_Whh  = (const float*)d_in[12];
    const float* agg_bih  = (const float*)d_in[13];
    const float* agg_bhh  = (const float*)d_in[14];
    const float* Wc1 = (const float*)d_in[15];
    const float* Wc2 = (const float*)d_in[16];
    const float* vc  = (const float*)d_in[17];
    const float* Wb  = (const float*)d_in[18];
    const float* Wd  = (const float*)d_in[19];
    const float* vd  = (const float*)d_in[20];
    const float* Wm  = (const float*)d_in[21];
    const float* vm  = (const float*)d_in[22];
    const float* Ws  = (const float*)d_in[23];
    const float* vs  = (const float*)d_in[24];
    const float* Wq  = (const float*)d_in[25];
    const float* vq  = (const float*)d_in[26];
    const float* Wp1 = (const float*)d_in[27];
    const float* Wp2 = (const float*)d_in[28];
    const float* vp  = (const float*)d_in[29];
    const float* Wpred = (const float*)d_in[30];
    float* out = (float*)d_out;

    // workspace carve-up (floats)
    float* ws = (float*)d_ws;
    size_t off = 0;
    auto alloc = [&](size_t n) { float* p = ws + off; off += n; return p; };
    float* xp      = alloc((size_t)B_ * N_ * E_);     // 307200
    float* xh      = alloc((size_t)B_ * M_ * E_);
    float* gxA     = alloc((size_t)B_ * T_ * G_);     // 460800
    float* gxB     = alloc((size_t)B_ * T_ * G_);
    float* hp      = alloc((size_t)B_ * N_ * D2_);
    float* hh      = alloc((size_t)B_ * M_ * D2_);
    float* agg     = alloc((size_t)B_ * M_ * AGG_);   // 1843200
    float* agg_rep = alloc((size_t)B_ * M_ * D2_);
    float* scores  = alloc((size_t)B_ * M_ * N_);
    float* tmpA    = alloc((size_t)B_ * N_ * D2_);    // reused: ac / hpWb / aq
    float* tmpC    = alloc((size_t)B_ * M_ * H_);     // reused: cc / cm
    float* t1      = alloc((size_t)B_ * M_ * H_);
    float* sq      = alloc((size_t)B_ * N_);
    float* rq      = alloc((size_t)B_ * D2_);
    float* rqp     = alloc((size_t)B_ * H_);
    float* sp      = alloc((size_t)B_ * M_);
    float* rp      = alloc((size_t)B_ * D2_);

    auto gemm = [&](const float* A, long long asb, int lda,
                    const float* Bw, long long bsb, int ldb,
                    const float* bias, float* C, long long csb, int ldc,
                    int M, int N, int K, int nb) {
        dim3 g((M + 127) / 128, (N + 63) / 64, nb);
        wmma_gemm_bias<<<g, 256, 0, stream>>>(A, asb, lda, Bw, bsb, ldb, bias, C, csb, ldc, M, N, K);
    };

    const size_t gruSmem = (size_t)G_ * H_ * 2 + 760 * 4;   // bf16 Whh + f32 scratch

    // 1) embedding gather
    gather_rows<<<B_ * N_, 128, 0, stream>>>(premise, emb, xp, B_ * N_, E_);
    gather_rows<<<B_ * M_, 128, 0, stream>>>(hypothesis, emb, xh, B_ * M_, E_);

    // 2) premise biGRU
    gemm(xp, 0, E_, prem_Wih,          0, E_, prem_bih,        gxA, 0, G_, B_ * N_, G_, E_, 1);
    gemm(xp, 0, E_, prem_Wih + G_ * E_, 0, E_, prem_bih + G_,  gxB, 0, G_, B_ * N_, G_, E_, 1);
    gru_scan<<<dim3(B_, 2), 512, gruSmem, stream>>>(gxA, gxB, prem_Whh, prem_bhh, hp, D2_);

    // 3) hypothesis biGRU
    gemm(xh, 0, E_, hypo_Wih,          0, E_, hypo_bih,        gxA, 0, G_, B_ * M_, G_, E_, 1);
    gemm(xh, 0, E_, hypo_Wih + G_ * E_, 0, E_, hypo_bih + G_,  gxB, 0, G_, B_ * M_, G_, E_, 1);
    gru_scan<<<dim3(B_, 2), 512, gruSmem, stream>>>(gxA, gxB, hypo_Whh, hypo_bhh, hh, D2_);

    // agg layout: [hh | qts | qtc | qtd | qtb | qtm], each 300 wide
    copy_to_agg<<<(B_ * M_ * D2_ + 255) / 256, 256, 0, stream>>>(hh, agg, (long long)B_ * M_ * D2_, D2_, AGG_, 0);

    // 4) qtc: concat attention
    gemm(hp, 0, D2_, Wc1, 0, D2_, nullptr, tmpA, 0, H_, B_ * N_, H_, D2_, 1);
    gemm(hh, 0, D2_, Wc2, 0, D2_, nullptr, tmpC, 0, H_, B_ * M_, H_, D2_, 1);
    add_attn_scores<<<dim3(M_, B_), 128, 0, stream>>>(tmpA, tmpC, vc, 1.0f, scores);
    softmax_attend<<<dim3(M_, B_), 256, 0, stream>>>(scores, hp, N_, D2_, agg, AGG_, (long long)M_ * AGG_, 2 * D2_);

    // 5) qtb: bilinear attention
    gemm(hp, 0, D2_, Wb, 0, D2_, nullptr, tmpA, 0, D2_, B_ * N_, D2_, D2_, 1);
    gemm(hh, (long long)M_ * D2_, D2_, tmpA, (long long)N_ * D2_, D2_, nullptr,
         scores, (long long)M_ * N_, N_, M_, N_, D2_, B_);
    softmax_attend<<<dim3(M_, B_), 256, 0, stream>>>(scores, hp, N_, D2_, agg, AGG_, (long long)M_ * AGG_, 4 * D2_);

    // 6) qtd: dot-trilinear attention
    trilinear_scores<<<dim3(M_, B_), 256, 0, stream>>>(hp, hh, Wd, vd, scores);
    softmax_attend<<<dim3(M_, B_), 256, 0, stream>>>(scores, hp, N_, D2_, agg, AGG_, (long long)M_ * AGG_, 3 * D2_);

    // 7) qtm: minus attention
    gemm(hp, 0, D2_, Wm, 0, D2_, nullptr, tmpA, 0, H_, B_ * N_, H_, D2_, 1);
    gemm(hh, 0, D2_, Wm, 0, D2_, nullptr, tmpC, 0, H_, B_ * M_, H_, D2_, 1);
    add_attn_scores<<<dim3(M_, B_), 128, 0, stream>>>(tmpA, tmpC, vm, -1.0f, scores);
    softmax_attend<<<dim3(M_, B_), 256, 0, stream>>>(scores, hp, N_, D2_, agg, AGG_, (long long)M_ * AGG_, 5 * D2_);

    // 8) qts: self trilinear attention (hh with hh)
    trilinear_scores<<<dim3(M_, B_), 256, 0, stream>>>(hh, hh, Ws, vs, scores);
    softmax_attend<<<dim3(M_, B_), 256, 0, stream>>>(scores, hh, M_, D2_, agg, AGG_, (long long)M_ * AGG_, 1 * D2_);

    // 9) aggregation biGRU over 1800-wide concat
    gemm(agg, 0, AGG_, agg_Wih,            0, AGG_, agg_bih,       gxA, 0, G_, B_ * M_, G_, AGG_, 1);
    gemm(agg, 0, AGG_, agg_Wih + G_ * AGG_, 0, AGG_, agg_bih + G_, gxB, 0, G_, B_ * M_, G_, AGG_, 1);
    gru_scan<<<dim3(B_, 2), 512, gruSmem, stream>>>(gxA, gxB, agg_Whh, agg_bhh, agg_rep, D2_);

    // 10) premise pooling: rq
    gemm(hp, 0, D2_, Wq, 0, D2_, nullptr, tmpA, 0, H_, B_ * N_, H_, D2_, 1);
    tanh_dot_v<<<(B_ * N_ + 127) / 128, 128, 0, stream>>>(tmpA, vq, sq, B_ * N_);
    softmax_attend<<<dim3(1, B_), 256, 0, stream>>>(sq, hp, N_, D2_, rq, 0, D2_, 0);

    // 11) agg pooling: rp
    gemm(agg_rep, 0, D2_, Wp1, 0, D2_, nullptr, t1, 0, H_, B_ * M_, H_, D2_, 1);
    gemm(rq, 0, D2_, Wp2, 0, D2_, nullptr, rqp, 0, H_, B_, H_, D2_, 1);
    sp_scores<<<(B_ * M_ + 255) / 256, 256, 0, stream>>>(t1, rqp, vp, sp);
    softmax_attend<<<dim3(1, B_), 256, 0, stream>>>(sp, agg_rep, M_, D2_, rp, 0, D2_, 0);

    // 12) prediction
    pred_kernel<<<B_, 32, 0, stream>>>(rp, Wpred, out);
}
